// SDF_37142877176626
// MI455X (gfx1250) — matrix operand order repaired and checked by reading, not compiled, not run on hardware
//
#include <hip/hip_runtime.h>
#include <hip/hip_bf16.h>

typedef _Float16 v16h __attribute__((ext_vector_type(16)));
typedef float    v8f  __attribute__((ext_vector_type(8)));

#define TBL (1u << 19)
#define WAVES 8

union U16H { v16h v; unsigned u[8]; };
union UPK  { _Float16 h[2]; unsigned u; };

// softplus(10x)/10 via raw base-2 HW transcendentals: 1+2^t is in (1,2], so no
// denorm/range fixup code is needed (unlike __logf/__expf lowering).
__device__ __forceinline__ float softplus10(float v) {
  float y = 10.0f * v;
  float t = __builtin_amdgcn_exp2f(-1.4426950408889634f * fabsf(y)); // 2^(-|y|*log2e)
  float l = __builtin_amdgcn_logf(1.0f + t);                         // log2(1+t)
  return 0.1f * fmaf(l, 0.69314718055994531f, fmaxf(y, 0.0f));
}

// A-matrix fragment (16x32 f16) from wave-private LDS tile (row = 64 halfs).
// Lanes 0-15: M=lane, K={0..15}; lanes 16-31: M=lane-16, K={16..31}.
// memcpy => aliasing-safe ds_load_b128 pairs that depend on the f16 act stores.
__device__ __forceinline__ v16h ld_a(const _Float16* hrow, int lane, int kt) {
  U16H r; int M = lane & 15, hi = lane >> 4;
  int K0 = kt * 32 + hi * 16;
  const char* base = (const char*)(hrow + M * 64);
  __builtin_memcpy(&r.u[0], base + (size_t)K0 * 2, 16);
  __builtin_memcpy(&r.u[4], base + (size_t)(K0 + 8) * 2, 16);
  return r.v;
}

// B-matrix fragment (32x16 f16): B[k][n] = W[nt*16+n][kt*32+k]; W row-major [out][kdim].
__device__ __forceinline__ v16h ld_b(const _Float16* w, int kdim, int lane, int nt, int kt) {
  U16H r; int N = lane & 15, hi = lane >> 4;
  const char* base = (const char*)(w + (nt * 16 + N) * kdim + kt * 32 + hi * 16);
  __builtin_memcpy(&r.u[0], base, 16);
  __builtin_memcpy(&r.u[4], base + 16, 16);
  return r.v;
}

// Last layer: B has W4 in column 0, zeros elsewhere.
__device__ __forceinline__ v16h ld_b4(const _Float16* w4, int lane, int kt) {
  U16H r; int N = lane & 15, hi = lane >> 4;
#pragma unroll
  for (int v = 0; v < 8; ++v) {
    unsigned val;
    __builtin_memcpy(&val, (const char*)w4 + (size_t)(kt * 16 + hi * 8 + v) * 4, 4);
    r.u[v] = (N == 0) ? val : 0u;
  }
  return r.v;
}

// D layout: lane l, reg r -> (M = (l>>4)*8 + r, N = l&15). Activate + f16 store (typed).
__device__ __forceinline__ void act_store(_Float16* hrow, int lane, int nt, v8f acc) {
  int N = lane & 15, hi = lane >> 4;
#pragma unroll
  for (int r = 0; r < 8; ++r)
    hrow[(hi * 8 + r) * 64 + nt * 16 + N] = (_Float16)softplus10(acc[r]);
}

__global__ __launch_bounds__(256) void sdf_fused_kernel(
    const float* __restrict__ x, const float* __restrict__ grid,
    const float* __restrict__ W1, const float* __restrict__ W2,
    const float* __restrict__ W3, const float* __restrict__ W4,
    float* __restrict__ out, int ntiles)
{
  __shared__ __align__(16) _Float16 s_w1[64 * 32];
  __shared__ __align__(16) _Float16 s_w2[64 * 64];
  __shared__ __align__(16) _Float16 s_w3[64 * 64];
  __shared__ __align__(16) _Float16 s_w4[64];
  __shared__ __align__(16) _Float16 s_h[WAVES][16 * 64];   // per-wave activation tile

  const int tid = threadIdx.x;
  for (int i = tid; i < 64 * 32; i += 256) s_w1[i] = (_Float16)W1[i];
  for (int i = tid; i < 64 * 64; i += 256) s_w2[i] = (_Float16)W2[i];
  for (int i = tid; i < 64 * 64; i += 256) s_w3[i] = (_Float16)W3[i];
  if (tid < 64) s_w4[tid] = (_Float16)W4[tid];
  __syncthreads();

  const int lane = tid & 31;
  const int wid  = tid >> 5;
  _Float16* hrow = &s_h[wid][0];

  const v8f zac = {0.f, 0.f, 0.f, 0.f, 0.f, 0.f, 0.f, 0.f};
  const int gwave = blockIdx.x * WAVES + wid;
  const int nwave = gridDim.x * WAVES;
  const int p = lane & 15;      // point within 16-point tile
  const int lhalf = lane >> 4;  // lane-half selects level parity / M-half

  for (int tile = gwave; tile < ntiles; tile += nwave) {
    // Opaque zero offset: keeps per-layer weight-fragment DS loads inside the
    // loop (defeats LICM re-hoisting them into 176 pinned VGPRs).
    unsigned zoff = 0;
    asm volatile("" : "+v"(zoff));

    // ---------- hash-grid encode: 16 points x 16 levels over 32 lanes ----------
    const float* xp = x + (size_t)(tile * 16 + p) * 3;
    __builtin_prefetch(x + (size_t)((tile + nwave) * 16 + p) * 3, 0, 0);
    float x0 = xp[0], x1 = xp[1], x2 = xp[2];
#pragma unroll 2
    for (int i = 0; i < 8; ++i) {
      int level = lhalf + 2 * i;
      float scale = exp2f(4.0f + (7.0f / 15.0f) * (float)level) - 1.0f; // 16*B^l - 1
      float px = x0 * scale + 0.5f, py = x1 * scale + 0.5f, pz = x2 * scale + 0.5f;
      float fx = floorf(px), fy = floorf(py), fz = floorf(pz);
      float rx = px - fx, ry = py - fy, rz = pz - fz;
      int gx = (int)fx, gy = (int)fy, gz = (int)fz;
      int res = (int)ceilf(scale) + 1;
      bool dense = (res <= 80);                 // res^3 <= 2^19
      const float2* gl = ((const float2*)grid) + (size_t)level * TBL;
      float f0 = 0.f, f1 = 0.f;
#pragma unroll
      for (int c = 0; c < 8; ++c) {
        int cx = c & 1, cy = (c >> 1) & 1, cz = (c >> 2) & 1;
        float w = (cx ? rx : 1.f - rx) * (cy ? ry : 1.f - ry) * (cz ? rz : 1.f - rz);
        int ix = gx + cx, iy = gy + cy, iz = gz + cz;
        unsigned idx;
        if (dense) {
          ix = ix < 0 ? 0 : (ix > res - 1 ? res - 1 : ix);
          iy = iy < 0 ? 0 : (iy > res - 1 ? res - 1 : iy);
          iz = iz < 0 ? 0 : (iz > res - 1 ? res - 1 : iz);
          idx = (unsigned)(ix + res * (iy + res * iz));
        } else {
          idx = ((unsigned)ix ^ (unsigned)iy * 2654435761u ^ (unsigned)iz * 805459861u)
                & (TBL - 1u);
        }
        float2 g = gl[idx];                     // L2-resident gather (64 MB tables)
        f0 = fmaf(w, g.x, f0);
        f1 = fmaf(w, g.y, f1);
      }
      UPK pk; pk.h[0] = (_Float16)f0; pk.h[1] = (_Float16)f1;
      __builtin_memcpy(hrow + p * 64 + level * 2, &pk.u, 4);  // enc[p][2l..2l+1]
    }
    // Wave-private tile + in-order DS within a wave => no barrier needed.

    v8f acc[4];
    // ---------- layer 1: enc[16x32] @ W1^T -> h[16x64] ----------
    {
      v16h a0 = ld_a(hrow, lane, 0);
      v16h bb[4];
#pragma unroll
      for (int nt = 0; nt < 4; ++nt) bb[nt] = ld_b(s_w1 + zoff, 32, lane, nt, 0);
#pragma unroll
      for (int nt = 0; nt < 4; ++nt)
        acc[nt] = __builtin_amdgcn_wmma_f32_16x16x32_f16(false, a0, false, bb[nt],
                                                         (short)0, zac, false, false);
#pragma unroll
      for (int nt = 0; nt < 4; ++nt) act_store(hrow, lane, nt, acc[nt]);
    }
    // ---------- layer 2: h[16x64] @ W2^T ----------
    {
      v16h a0 = ld_a(hrow, lane, 0), a1 = ld_a(hrow, lane, 1);
      v16h c0[4], c1[4];
#pragma unroll
      for (int nt = 0; nt < 4; ++nt) {
        c0[nt] = ld_b(s_w2 + zoff, 64, lane, nt, 0);
        c1[nt] = ld_b(s_w2 + zoff, 64, lane, nt, 1);
      }
#pragma unroll
      for (int nt = 0; nt < 4; ++nt)
        acc[nt] = __builtin_amdgcn_wmma_f32_16x16x32_f16(false, a0, false, c0[nt],
                                                         (short)0, zac, false, false);
#pragma unroll
      for (int nt = 0; nt < 4; ++nt)
        acc[nt] = __builtin_amdgcn_wmma_f32_16x16x32_f16(false, a1, false, c1[nt],
                                                         (short)0, acc[nt], false, false);
#pragma unroll
      for (int nt = 0; nt < 4; ++nt) act_store(hrow, lane, nt, acc[nt]);
    }
    // ---------- layer 3 ----------
    {
      v16h a0 = ld_a(hrow, lane, 0), a1 = ld_a(hrow, lane, 1);
      v16h c0[4], c1[4];
#pragma unroll
      for (int nt = 0; nt < 4; ++nt) {
        c0[nt] = ld_b(s_w3 + zoff, 64, lane, nt, 0);
        c1[nt] = ld_b(s_w3 + zoff, 64, lane, nt, 1);
      }
#pragma unroll
      for (int nt = 0; nt < 4; ++nt)
        acc[nt] = __builtin_amdgcn_wmma_f32_16x16x32_f16(false, a0, false, c0[nt],
                                                         (short)0, zac, false, false);
#pragma unroll
      for (int nt = 0; nt < 4; ++nt)
        acc[nt] = __builtin_amdgcn_wmma_f32_16x16x32_f16(false, a1, false, c1[nt],
                                                         (short)0, acc[nt], false, false);
#pragma unroll
      for (int nt = 0; nt < 4; ++nt) act_store(hrow, lane, nt, acc[nt]);
    }
    // ---------- layer 4: h[16x64] @ W4 (col 0 of B) ----------
    {
      v16h a0 = ld_a(hrow, lane, 0), a1 = ld_a(hrow, lane, 1);
      v16h d0 = ld_b4(s_w4 + zoff, lane, 0), d1 = ld_b4(s_w4 + zoff, lane, 1);
      v8f r = __builtin_amdgcn_wmma_f32_16x16x32_f16(false, a0, false, d0,
                                                     (short)0, zac, false, false);
      r = __builtin_amdgcn_wmma_f32_16x16x32_f16(false, a1, false, d1,
                                                 (short)0, r, false, false);
      if (p == 0) {                             // lanes 0 & 16 hold column 0 of D
        int base = tile * 16 + lhalf * 8;
#pragma unroll
        for (int rr = 0; rr < 8; ++rr) out[base + rr] = r[rr];
      }
    }
  }
}

extern "C" void kernel_launch(void* const* d_in, const int* in_sizes, int n_in,
                              void* d_out, int out_size, void* d_ws, size_t ws_size,
                              hipStream_t stream) {
  (void)n_in; (void)d_ws; (void)ws_size; (void)out_size;
  const float* x    = (const float*)d_in[0];
  const float* grid = (const float*)d_in[1];
  const float* W1   = (const float*)d_in[2];
  const float* W2   = (const float*)d_in[3];
  const float* W3   = (const float*)d_in[4];
  const float* W4   = (const float*)d_in[5];
  float* out = (float*)d_out;

  int npts   = in_sizes[0] / 3;   // 2^20
  int ntiles = npts / 16;         // 65536 16-point tiles
  int blocks = 1024;              // 8192 persistent waves, ~8 tiles each
  sdf_fused_kernel<<<blocks, 256, 0, stream>>>(x, grid, W1, W2, W3, W4, out, ntiles);
}
